// TacStModel_42451456753666
// MI455X (gfx1250) — matrix operand order repaired and checked by reading, not compiled, not run on hardware
//
#include <hip/hip_runtime.h>
#include <hip/hip_bf16.h>

// ---- problem constants (from reference) ----
#define BATCH 256
#define LSEQ  128
#define ST    128      // STATE
#define KDIM  256      // 2*STATE (GEMM K)
#define GATES 640      // 5*STATE
#define OUTD  3

typedef __attribute__((ext_vector_type(16))) __bf16 v16bf;
typedef __attribute__((ext_vector_type(8)))  __bf16 v8bf;
typedef __attribute__((ext_vector_type(8)))  float  v8f;
typedef __attribute__((ext_vector_type(4)))  float  v4f;

__device__ __forceinline__ float sgm(float x) {
    return 1.0f / (1.0f + __expf(-x));
}

// ---------------------------------------------------------------------------
// Kernel 1: convert whx_w (f32 [640][256], row = gate output, col = K) to a
// bf16 copy in workspace. Row-major layout means each WMMA B-tile column is a
// contiguous 32-byte run -> one b128 pair per lane. 327,680 B, L2-resident.
// ---------------------------------------------------------------------------
__global__ void cvt_w_kernel(const float* __restrict__ w, __bf16* __restrict__ wb) {
    int i = blockIdx.x * blockDim.x + threadIdx.x;
    if (i < GATES * KDIM) wb[i] = (__bf16)w[i];
}

// ---------------------------------------------------------------------------
// Kernel 2: full recurrence. grid = 16 WGs (16 batch rows each), 256 thr/WG
// (8 waves). Wave w owns output columns [16w,16w+16) of each of the 5 gates.
// ---------------------------------------------------------------------------
__launch_bounds__(256)
__global__ void treelstm_kernel(const int*    __restrict__ ids,
                                const float*  __restrict__ embed,
                                const __bf16* __restrict__ wb,
                                const float*  __restrict__ whx_b,
                                const float*  __restrict__ init_state,
                                const float*  __restrict__ final_w,
                                const float*  __restrict__ final_b,
                                float*        __restrict__ out)
{
    __shared__ __bf16 hA[2][16][KDIM];   // GEMM A operand: [left_h | right_h] bf16 (dbl-buf)
    __shared__ float  cL[16][ST];        // left cell state (f32, owner-updated)
    __shared__ float  xbuf[2][16][KDIM]; // x_t f32 = [right_h | right_c] (dbl-buf)
    __shared__ float  hF[16][ST];        // final-step h in f32 for output projection

    const int tid  = threadIdx.x;
    const int lane = tid & 31;
    const int wv   = tid >> 5;          // wave 0..7
    const int b0   = blockIdx.x * 16;   // batch tile base
    const int jn   = lane & 15;         // N within tile / A-row / C-col
    const int half = lane >> 4;         // K-half (A/B), M-half (C/D)
    const int j    = 16 * wv + jn;      // gate column owned for elementwise

    // x-prefetch mapping: thread -> (row pm, four 4-col groups at cols 4*pp + 64*q, q=0..3)
    const int pm = tid >> 4, pp = tid & 15;

    // ---- prologue: broadcast init_state, load x_0 ----
    for (int idx = tid; idx < 16 * ST; idx += 256) {
        int m = idx >> 7, jj = idx & 127;
        cL[m][jj]    = init_state[ST + jj];       // left_c
        hA[0][m][jj] = (__bf16)init_state[jj];    // left_h
    }
    {
        int id0 = ids[(b0 + pm) * LSEQ + 0];
        const float* src = embed + (size_t)id0 * KDIM;
        #pragma unroll
        for (int q = 0; q < 4; ++q) {
            int c0 = 4 * (pp + 16 * q);
            v4f v = *(const v4f*)(src + c0);
            *(v4f*)&xbuf[0][pm][c0] = v;
            if (c0 < ST) {                        // right_h -> A-operand upper K half
                hA[0][pm][ST + c0 + 0] = (__bf16)v.x;
                hA[0][pm][ST + c0 + 1] = (__bf16)v.y;
                hA[0][pm][ST + c0 + 2] = (__bf16)v.z;
                hA[0][pm][ST + c0 + 3] = (__bf16)v.w;
            }
        }
    }
    __syncthreads();

    // loop-invariant per-lane gate biases
    float bias[5];
    #pragma unroll
    for (int g = 0; g < 5; ++g) bias[g] = whx_b[g * ST + j];

    const v8f vzero = {0.f, 0.f, 0.f, 0.f, 0.f, 0.f, 0.f, 0.f};

    for (int t = 0; t < LSEQ; ++t) {
        const int cur = t & 1, nxt = cur ^ 1;

        // ---- issue x_{t+1} gather early (latency overlaps the GEMM) ----
        v4f xv[4];
        if (t + 1 < LSEQ) {
            int id = ids[(b0 + pm) * LSEQ + (t + 1)];
            const float* src = embed + (size_t)id * KDIM;
            #pragma unroll
            for (int q = 0; q < 4; ++q) xv[q] = *(const v4f*)(src + 4 * (pp + 16 * q));
        }

        // ---- GEMM: 5 accum tiles, K=256 in 8 chunks of 32 ----
        v8f acc[5];
        #pragma unroll
        for (int g = 0; g < 5; ++g) acc[g] = vzero;

        #pragma unroll
        for (int kt = 0; kt < 8; ++kt) {
            const int k0 = 32 * kt;
            // A 16x32 bf16: lane holds row jn; 16 contiguous K starting k0+16*half
            v8bf a0 = *(const v8bf*)&hA[cur][jn][k0 + 16 * half];
            v8bf a1 = *(const v8bf*)&hA[cur][jn][k0 + 16 * half + 8];
            v16bf a;
            #pragma unroll
            for (int i = 0; i < 8; ++i) { a[i] = a0[i]; a[8 + i] = a1[i]; }
            #pragma unroll
            for (int g = 0; g < 5; ++g) {
                // B 32x16 bf16: lane holds column n; 16 contiguous K starting k0+16*half
                const int n = g * ST + j;
                v16bf b = *(const v16bf*)&wb[(size_t)n * KDIM + k0 + 16 * half];
                acc[g] = __builtin_amdgcn_wmma_f32_16x16x32_bf16(
                             false, a, false, b, (short)0, acc[g], false, false);
            }
        }

        // ---- elementwise gates; this wave owns column j for all 16 rows ----
        #pragma unroll
        for (int r = 0; r < 8; ++r) {
            const int m = r + 8 * half;           // C/D row mapping
            float av = acc[0][r] + bias[0];
            float iv = acc[1][r] + bias[1];
            float f1 = acc[2][r] + bias[2];
            float f2 = acc[3][r] + bias[3];
            float ov = acc[4][r] + bias[4];
            float lc = cL[m][j];
            float rc = xbuf[cur][m][ST + j];      // right_c = x_t[:,128:]
            float c  = tanhf(av) * sgm(iv) + sgm(f1) * lc + sgm(f2) * rc;
            float h  = sgm(ov) * tanhf(c);
            cL[m][j]      = c;
            hA[nxt][m][j] = (__bf16)h;            // left_h for next step
            if (t == LSEQ - 1) hF[m][j] = h;
        }

        // ---- commit prefetched x_{t+1} to the next buffers ----
        if (t + 1 < LSEQ) {
            #pragma unroll
            for (int q = 0; q < 4; ++q) {
                int c0 = 4 * (pp + 16 * q);
                *(v4f*)&xbuf[nxt][pm][c0] = xv[q];
                if (c0 < ST) {
                    hA[nxt][pm][ST + c0 + 0] = (__bf16)xv[q].x;
                    hA[nxt][pm][ST + c0 + 1] = (__bf16)xv[q].y;
                    hA[nxt][pm][ST + c0 + 2] = (__bf16)xv[q].z;
                    hA[nxt][pm][ST + c0 + 3] = (__bf16)xv[q].w;
                }
            }
        }
        __syncthreads();   // single barrier per step (double-buffered LDS)
    }

    // ---- final projection: out[m,o] = h . final_w[o] + final_b[o] (tiny) ----
    if (tid < 16 * OUTD) {
        int m = tid / OUTD, o = tid % OUTD;
        float s = final_b[o];
        #pragma unroll 4
        for (int k = 0; k < ST; ++k) s += hF[m][k] * final_w[o * ST + k];
        out[(b0 + m) * OUTD + o] = s;
    }
}

// ---------------------------------------------------------------------------
extern "C" void kernel_launch(void* const* d_in, const int* in_sizes, int n_in,
                              void* d_out, int out_size, void* d_ws, size_t ws_size,
                              hipStream_t stream) {
    const int*   ids        = (const int*)  d_in[0];
    const float* embed      = (const float*)d_in[1];
    const float* whx_w      = (const float*)d_in[2];
    const float* whx_b      = (const float*)d_in[3];
    const float* init_state = (const float*)d_in[4];
    const float* final_w    = (const float*)d_in[5];
    const float* final_b    = (const float*)d_in[6];
    float*       out        = (float*)d_out;

    __bf16* wb = (__bf16*)d_ws;   // 640*256*2 = 327,680 bytes of workspace

    cvt_w_kernel<<<(GATES * KDIM + 255) / 256, 256, 0, stream>>>(whx_w, wb);
    treelstm_kernel<<<dim3(BATCH / 16), dim3(256), 0, stream>>>(
        ids, embed, wb, whx_b, init_state, final_w, final_b, out);
}
